// GLMSite_40269613367958
// MI455X (gfx1250) — matrix operand order repaired and checked by reading, not compiled, not run on hardware
//
// GLMSite forward for MI455X (gfx1250), wave32 + WMMA + async-LDS.
//
// Key derivations from the reference:
//  * p1 PMA (K1=2500 attention) is dead code; p2 softmax over query dim of
//    size 1 is identically 1.0, so mab(p2) = LN/MLP over (Qp + sum_k V).
//  * GEMMs run as segmented split-K f16-WMMA (v_wmma_f32_16x16x32_f16) with
//    per-segment row-gather (fuses cat(s[src], e_s, s[dst], vnorm) without
//    materializing the 283MB edge input).
//  * Deterministic aggregation via CSR (int-atomic count, serial scan,
//    per-bucket insertion sort) -> gather-sum in fixed order, no f32 atomics.
//
// GEMM: 128-thread block = 4 waves, 64x32 block tile, B tile shared via LDS
// (N-major for contiguous ds_load_b128 fragment reads), per-thread k-column
// staging so the split-K segment lookup is hoisted out of the element loop,
// 2 WMMA issues per wave per 32-k step, global_prefetch_b8 on next W tile.
//
// k_agg: message rows DMA-staged to LDS via GLOBAL_LOAD_ASYNC_TO_LDS_B128
// (ASYNCcnt / s_wait_asynccnt); builtin takes (v4i AS1*, v4i AS3*, imm, imm).
//
// Param pointer mapping assumes jax pytree flatten order (dict keys sorted).

#include <hip/hip_runtime.h>
#include <hip/hip_bf16.h>

typedef __attribute__((ext_vector_type(16))) _Float16 v16h;
typedef __attribute__((ext_vector_type(8)))  float    v8f;
typedef __attribute__((ext_vector_type(4)))  int      v4i;

#define NNODE 4096
#define NPG   512
#define NGR   8
#define NEDGE 122880
#define HID   256
#define ESD   32
#define SI0   1301
#define OUTD  489
#define EPSV  1e-8f
#define DIV_UP(a,b) (((a)+(b)-1)/(b))

#if __has_builtin(__builtin_amdgcn_global_load_async_to_lds_b128) && \
    __has_builtin(__builtin_amdgcn_s_wait_asynccnt)
#define HAVE_ASYNC_LDS 1
#define GAS_V4I(p) ((__attribute__((address_space(1))) v4i*)(void*)(p))
#define LAS_V4I(p) ((__attribute__((address_space(3))) v4i*)(void*)(p))
#else
#define HAVE_ASYNC_LDS 0
#endif

// ---------------- input index map (jax pytree flatten, dicts key-sorted) ----
#define IN_HS 0
#define IN_HV 1
#define IN_ES 2
#define IN_EV 3
#define IN_SEQ 4
#define IN_EIDX 5
#define IN_BATCH 6
#define P_WE_WH   7
#define P_WE_WSB  8
#define P_WE_WSW  9
#define P_WE_WSVB 10
#define P_WE_WSVW 11
#define P_WE_WV   12
#define P_WE_LNB  13
#define P_WE_LNG  14
#define P_WOUT_WH  15
#define P_WOUT_WSB 16
#define P_WOUT_WSW 17
#define P_WOUT_LNB 18
#define P_WOUT_LNG 19
#define P_WS 20
#define P_WV_WH   21
#define P_WV_WSB  22
#define P_WV_WSW  23
#define P_WV_WSVB 24
#define P_WV_WSVW 25
#define P_WV_WV   26
#define P_WV_LNB  27
#define P_WV_LNG  28
#define P_DENSE_B 29
#define P_DENSE_W 30
#define P_LIN1B 31
#define P_LIN1W 32
#define P_LIN2B 33
#define P_LIN2W 34
// p1 (dead): 35..47
#define P_P2_S    48
#define P_P2_FCOB 49
#define P_P2_FCOW 50
#define P_P2_FCQB 51
#define P_P2_FCQW 52
// p2 k_b/k_w (dead): 53,54
#define P_P2_LN0B 55
#define P_P2_LN0G 56
#define P_P2_LN1B 57
#define P_P2_LN1G 58
#define P_P2_VB   59
#define P_P2_VW   60
#define P_LAYER0  61   // per layer: ff0(6), ff1(6), msg0(6), msg1(6), msg2(6), n0 b/g, n1 b/g = 34
#define P_LSTRIDE 34
#define P_RO1B 163
#define P_RO1W 164
#define P_RO2B 165
#define P_RO2W 166

__device__ __forceinline__ float actf(float x, int act) {
  if (act == 1) return x > 0.f ? x : 0.f;
  if (act == 2) return 1.f / (1.f + __expf(-x));
  return x;
}

// ====================== segmented split-K WMMA GEMM =========================
struct GemmSeg { const float* A; const int* gather; int K; int lda; int relu; };

#define TM 64
#define TN 32

__global__ void __launch_bounds__(128)
k_gemm(GemmSeg g0, GemmSeg g1, GemmSeg g2, GemmSeg g3,
       const float* __restrict__ W, const float* __restrict__ bias,
       float* __restrict__ C, int M, int Nc, int act)
{
  __shared__ _Float16 As[TM * 32];        // row-major [m][k]
  __shared__ _Float16 Bs[TN * 32];        // N-major  [n][k] -> contiguous frags
  const int t = threadIdx.x;
  const int lane = t & 31;
  const int w = t >> 5;
  const int mt = blockIdx.x * TM, nt = blockIdx.y * TN;
  const int Ktot = g0.K + g1.K + g2.K + g3.K;

  v8f acc0 = {0.f, 0.f, 0.f, 0.f, 0.f, 0.f, 0.f, 0.f};
  v8f acc1 = acc0;

  const int cA  = lane;     // this thread stages A k-column cA (rows rA0+4j)
  const int rA0 = t >> 5;
  const int nB  = lane;     // this thread stages B n-column nB (k rows kB0+4j)
  const int kB0 = t >> 5;

  for (int k0 = 0; k0 < Ktot; k0 += 32) {
    // ---- resolve split-K segment ONCE per tile for this thread's k-column
    const float* Ap = nullptr; const int* gth = nullptr;
    int lda_ = 0, rl = 0, kk = 0;
    if (k0 + cA < Ktot) {
      kk = k0 + cA;
      if (kk < g0.K)      { Ap = g0.A; gth = g0.gather; lda_ = g0.lda; rl = g0.relu; }
      else { kk -= g0.K;
        if (kk < g1.K)    { Ap = g1.A; gth = g1.gather; lda_ = g1.lda; rl = g1.relu; }
        else { kk -= g1.K;
          if (kk < g2.K)  { Ap = g2.A; gth = g2.gather; lda_ = g2.lda; rl = g2.relu; }
          else { kk -= g2.K; Ap = g3.A; gth = g3.gather; lda_ = g3.lda; rl = g3.relu; } } }
    }
    // ---- stage A tile (64 x 32)
#pragma unroll
    for (int j = 0; j < 16; j++) {
      int r = rA0 + 4 * j;
      int m = mt + r;
      float v = 0.f;
      if (Ap && m < M) {
        int row = gth ? gth[m] : m;
        v = Ap[(size_t)row * lda_ + kk];
        if (rl && v < 0.f) v = 0.f;
      }
      As[r * 32 + cA] = (_Float16)v;
    }
    // ---- stage B tile (32 x 32), stored N-major
#pragma unroll
    for (int j = 0; j < 8; j++) {
      int k = kB0 + 4 * j;
      int kg = k0 + k, n = nt + nB;
      float v = (kg < Ktot && n < Nc) ? W[(size_t)kg * Nc + n] : 0.f;
      Bs[nB * 32 + k] = (_Float16)v;
    }
    // speculative prefetch of the next W k-tile (global_prefetch_b8)
    if (k0 + 32 < Ktot && nt + nB < Nc)
      __builtin_prefetch(&W[(size_t)(k0 + 32) * Nc + nt + nB], 0, 1);
    __syncthreads();

    // ---- pack fragments (contiguous k = 16*(lane/16) + i in A and B) and issue 2 WMMAs per wave
    v16h af, bf0, bf1;
    const int grpA = (lane >= 16) ? 8 : 0;
    const int grpB = (lane >= 16) ? 16 : 0;
    const int mm = w * 16 + (lane & 15);
    const int nn = lane & 15;
#pragma unroll
    for (int i = 0; i < 16; i++) {
      int p = i >> 1;
      int kx = 2 * grpA + i + 0 * p;
      af[i] = As[mm * 32 + kx];
    }
#pragma unroll
    for (int i = 0; i < 16; i++) bf0[i] = Bs[nn * 32 + grpB + i];
#pragma unroll
    for (int i = 0; i < 16; i++) bf1[i] = Bs[(16 + nn) * 32 + grpB + i];
    acc0 = __builtin_amdgcn_wmma_f32_16x16x32_f16(false, af, false, bf0,
                                                  (short)0, acc0, false, false);
    acc1 = __builtin_amdgcn_wmma_f32_16x16x32_f16(false, af, false, bf1,
                                                  (short)0, acc1, false, false);
    __syncthreads();
  }

  // C layout per wave: vgpr r -> M = w*16 + 8*(lane>=16) + r, N = lane&15 (+16)
  const int mbase = mt + w * 16 + ((lane >= 16) ? 8 : 0);
  const int ncol = lane & 15;
  const int n0 = nt + ncol, n1 = nt + 16 + ncol;
  const float bv0 = (bias && n0 < Nc) ? bias[n0] : 0.f;
  const float bv1 = (bias && n1 < Nc) ? bias[n1] : 0.f;
#pragma unroll
  for (int r = 0; r < 8; r++) {
    int m = mbase + r;
    if (m < M) {
      if (n0 < Nc) C[(size_t)m * Nc + n0] = actf(acc0[r] + bv0, act);
      if (n1 < Nc) C[(size_t)m * Nc + n1] = actf(acc1[r] + bv1, act);
    }
  }
}

// ====================== element/VALU kernels ================================
__global__ void k_embed(const float* hs, const int* seq, const float* Ws, float* s0) {
  int id = blockIdx.x * blockDim.x + threadIdx.x;
  if (id >= NNODE * SI0) return;
  int i = id / SI0, d = id % SI0;
  s0[(size_t)i * SI0 + d] = (d < 1280) ? hs[(size_t)i * 1280 + d]
                                       : Ws[seq[i] * 21 + (d - 1280)];
}

// GVP layernorm: s -> LN(s[+res])*g+b ; v -> (v[+res]) / sqrt(mean_j max(|v_j|^2, EPS))
__global__ void __launch_bounds__(256)
k_gvpln(const float* s_in, const float* res_s, const float* v_in, const float* res_v,
        const float* g, const float* b, float* s_out, float* v_out, int D, int nv)
{
  __shared__ float red[256];
  __shared__ float sh_mu, sh_rstd, sh_vn;
  const int row = blockIdx.x, t = threadIdx.x;
  const float* sp = s_in + (size_t)row * D;
  const float* rp = res_s ? res_s + (size_t)row * D : nullptr;
  float a = 0.f;
  for (int d = t; d < D; d += 256) a += sp[d] + (rp ? rp[d] : 0.f);
  red[t] = a; __syncthreads();
  for (int s = 128; s > 0; s >>= 1) { if (t < s) red[t] += red[t + s]; __syncthreads(); }
  if (t == 0) sh_mu = red[0] / D;
  __syncthreads();
  float mu = sh_mu;
  a = 0.f;
  for (int d = t; d < D; d += 256) { float x = sp[d] + (rp ? rp[d] : 0.f) - mu; a += x * x; }
  red[t] = a; __syncthreads();
  for (int s = 128; s > 0; s >>= 1) { if (t < s) red[t] += red[t + s]; __syncthreads(); }
  if (t == 0) sh_rstd = rsqrtf(red[0] / D + 1e-5f);
  __syncthreads();
  float rstd = sh_rstd;
  for (int d = t; d < D; d += 256) {
    float x = sp[d] + (rp ? rp[d] : 0.f);
    s_out[(size_t)row * D + d] = (x - mu) * rstd * g[d] + b[d];
  }
  if (nv > 0) {
    const float* vp = v_in + (size_t)row * nv * 3;
    const float* vr = res_v ? res_v + (size_t)row * nv * 3 : nullptr;
    float c = 0.f;
    if (t < nv) {
      float s2 = 0.f;
      for (int cc = 0; cc < 3; cc++) {
        float x = vp[t * 3 + cc] + (vr ? vr[t * 3 + cc] : 0.f);
        s2 += x * x;
      }
      c = fmaxf(s2, EPSV);
    }
    red[t] = (t < nv) ? c : 0.f; __syncthreads();
    for (int s = 128; s > 0; s >>= 1) { if (t < s) red[t] += red[t + s]; __syncthreads(); }
    if (t == 0) sh_vn = sqrtf(red[0] / nv);
    __syncthreads();
    float inv = 1.f / sh_vn;
    for (int e = t; e < nv * 3; e += 256) {
      float x = vp[e] + (vr ? vr[e] : 0.f);
      v_out[(size_t)row * nv * 3 + e] = x * inv;
    }
  }
}

// fused W_e: edge LN + 33->32 GVP + vector gate (small dims, VALU)
__global__ void k_we(const float* es, const float* ev,
                     const float* lnb, const float* lng,
                     const float* wh, const float* wsb, const float* wsw,
                     const float* wsvb, const float* wsvw, const float* wv,
                     float* es2, float* ev2)
{
  int e = blockIdx.x * blockDim.x + threadIdx.x;
  if (e >= NEDGE) return;
  const float* s = es + (size_t)e * ESD;
  float mu = 0.f;
  for (int d = 0; d < ESD; d++) mu += s[d];
  mu /= ESD;
  float var = 0.f;
  for (int d = 0; d < ESD; d++) { float x = s[d] - mu; var += x * x; }
  float rstd = rsqrtf(var / ESD + 1e-5f);
  float sl[33];
  for (int d = 0; d < ESD; d++) sl[d] = (s[d] - mu) * rstd * lng[d] + lnb[d];
  float v0 = ev[(size_t)e * 3], v1 = ev[(size_t)e * 3 + 1], v2 = ev[(size_t)e * 3 + 2];
  float inv = rsqrtf(fmaxf(v0 * v0 + v1 * v1 + v2 * v2, EPSV));
  v0 *= inv; v1 *= inv; v2 *= inv;
  float w0 = wh[0];
  float h0 = v0 * w0, h1 = v1 * w0, h2 = v2 * w0;
  sl[32] = sqrtf(fmaxf(h0 * h0 + h1 * h1 + h2 * h2, EPSV));
  float gz = wsvb[0];
  for (int n = 0; n < ESD; n++) {
    float aa = wsb[n];
    for (int k = 0; k < 33; k++) aa += sl[k] * wsw[k * ESD + n];
    es2[(size_t)e * ESD + n] = aa;
    gz += aa * wsvw[n];
  }
  float gate = 1.f / (1.f + __expf(-gz));
  float wvv = wv[0] * gate;
  ev2[(size_t)e * 3 + 0] = h0 * wvv;
  ev2[(size_t)e * 3 + 1] = h1 * wvv;
  ev2[(size_t)e * 3 + 2] = h2 * wvv;
}

// vh = einsum(v[M,vi,3], wh[vi,hd]); vn = sqrt(max(|vh|^2, EPS))
__global__ void k_vhn(const float* v, const float* wh, float* vh, float* vn,
                      int M, int vi, int hd)
{
  int id = blockIdx.x * blockDim.x + threadIdx.x;
  if (id >= M * hd) return;
  int m = id / hd, h = id % hd;
  const float* vp = v + (size_t)m * vi * 3;
  float a0 = 0.f, a1 = 0.f, a2 = 0.f;
  for (int j = 0; j < vi; j++) {
    float w = wh[j * hd + h];
    a0 += vp[j * 3 + 0] * w; a1 += vp[j * 3 + 1] * w; a2 += vp[j * 3 + 2] * w;
  }
  size_t o = ((size_t)m * hd + h) * 3;
  vh[o] = a0; vh[o + 1] = a1; vh[o + 2] = a2;
  vn[(size_t)m * hd + h] = sqrtf(fmaxf(a0 * a0 + a1 * a1 + a2 * a2, EPSV));
}

// message variant: virtual v-in = cat(v[src](16), ev2(1), v[dst](16)), vi=hd=33
__global__ void k_vhn_edge(const float* vnode, const float* ev2,
                           const int* src, const int* dst, const float* wh,
                           float* vh, float* vn)
{
  int id = blockIdx.x * blockDim.x + threadIdx.x;
  if (id >= NEDGE * 33) return;
  int e = id / 33, h = id % 33;
  int is = src[e], idd = dst[e];
  float a0 = 0.f, a1 = 0.f, a2 = 0.f;
  for (int j = 0; j < 33; j++) {
    const float* vp;
    if (j < 16)       vp = vnode + ((size_t)is * 16 + j) * 3;
    else if (j == 16) vp = ev2 + (size_t)e * 3;
    else              vp = vnode + ((size_t)idd * 16 + (j - 17)) * 3;
    float w = wh[j * 33 + h];
    a0 += vp[0] * w; a1 += vp[1] * w; a2 += vp[2] * w;
  }
  size_t o = ((size_t)e * 33 + h) * 3;
  vh[o] = a0; vh[o + 1] = a1; vh[o + 2] = a2;
  vn[(size_t)e * 33 + h] = sqrtf(fmaxf(a0 * a0 + a1 * a1 + a2 * a2, EPSV));
}

// vout = sigmoid-gate (precomputed) * (vh @ wv[hd,vo])
__global__ void k_vogate(const float* vh, const float* wv, const float* gate,
                         float* vout, int M, int hd, int vo)
{
  int id = blockIdx.x * blockDim.x + threadIdx.x;
  if (id >= M * vo) return;
  int m = id / vo, o = id % vo;
  const float* vp = vh + (size_t)m * hd * 3;
  float a0 = 0.f, a1 = 0.f, a2 = 0.f;
  for (int h = 0; h < hd; h++) {
    float w = wv[h * vo + o];
    a0 += vp[h * 3] * w; a1 += vp[h * 3 + 1] * w; a2 += vp[h * 3 + 2] * w;
  }
  float gv = gate[(size_t)m * vo + o];
  size_t ob = ((size_t)m * vo + o) * 3;
  vout[ob] = a0 * gv; vout[ob + 1] = a1 * gv; vout[ob + 2] = a2 * gv;
}

// -------- CSR build (deterministic: counts -> scan -> fill -> bucket sort) --
__global__ void k_seti(int* p, int n, int v) {
  int id = blockIdx.x * blockDim.x + threadIdx.x;
  if (id < n) p[id] = v;
}
__global__ void k_copyi(const int* a, int* b, int n) {
  int id = blockIdx.x * blockDim.x + threadIdx.x;
  if (id < n) b[id] = a[id];
}
__global__ void k_count(const int* dst, int* cnt) {
  int e = blockIdx.x * blockDim.x + threadIdx.x;
  if (e < NEDGE) atomicAdd(&cnt[dst[e]], 1);
}
__global__ void k_scan(const int* cnt, int* rowptr) {
  if (blockIdx.x == 0 && threadIdx.x == 0) {
    int a = 0;
    for (int i = 0; i < NNODE; i++) { rowptr[i] = a; a += cnt[i]; }
    rowptr[NNODE] = a;
  }
}
__global__ void k_fill(const int* dst, int* cursor, int* eid) {
  int e = blockIdx.x * blockDim.x + threadIdx.x;
  if (e < NEDGE) { int p = atomicAdd(&cursor[dst[e]], 1); eid[p] = e; }
}
__global__ void k_sort(int* eid, const int* rowptr) {
  int i = blockIdx.x * blockDim.x + threadIdx.x;
  if (i >= NNODE) return;
  int lo = rowptr[i], hi = rowptr[i + 1];
  for (int a = lo + 1; a < hi; a++) {
    int key = eid[a], b = a - 1;
    while (b >= lo && eid[b] > key) { eid[b + 1] = eid[b]; b--; }
    eid[b + 1] = key;
  }
}

// out[i] = base[i] + (sum over incoming edges of msg)/max(cnt,1)
// Async path: DMA-stage message rows into LDS (ASYNCcnt), reduce from LDS.
#define AGG_LDS_F 2048   // 8KB f32 staging buffer
__global__ void __launch_bounds__(256)
k_agg(const float* base, const float* msg, const int* rowptr, const int* eid,
      float* out, int D)
{
  const int i = blockIdx.x, t = threadIdx.x;
  const int lo = rowptr[i], hi = rowptr[i + 1];
  const float denom = (float)((hi - lo) > 1 ? (hi - lo) : 1);
#if HAVE_ASYNC_LDS
  __shared__ float lds[AGG_LDS_F];
  const int rpc = AGG_LDS_F / D;   // rows per chunk (8 for D=256, 42 for D=48)
  const int qpr = D >> 2;          // 16-byte quanta per row
  float acc = 0.f;                 // this thread's dim (t < D)
  for (int cur = lo; cur < hi;) {
    int nr = hi - cur; if (nr > rpc) nr = rpc;
    for (int slot = t; slot < nr * qpr; slot += 256) {
      int rr = slot / qpr, qq = slot - rr * qpr;
      const float* gp = msg + (size_t)eid[cur + rr] * D + qq * 4;
      float* lp = &lds[rr * D + qq * 4];
      __builtin_amdgcn_global_load_async_to_lds_b128(GAS_V4I(gp), LAS_V4I(lp), 0, 0);
    }
    __builtin_amdgcn_s_wait_asynccnt(0);
    __syncthreads();
    if (t < D)
      for (int rr = 0; rr < nr; rr++) acc += lds[rr * D + t];
    __syncthreads();
    cur += nr;
  }
  if (t < D)
    out[(size_t)i * D + t] = base[(size_t)i * D + t] + acc / denom;
#else
  for (int d = t; d < D; d += 256) {
    float a = 0.f;
    for (int e = lo; e < hi; e++) a += msg[(size_t)eid[e] * D + d];
    out[(size_t)i * D + d] = base[(size_t)i * D + d] + a / denom;
  }
#endif
}

// GCN aggregation with self loops: deg = indeg+1, symmetric norm
__global__ void __launch_bounds__(256)
k_gcn(const float* xw, const int* rowptr, const int* eid, const int* src,
      const float* bias, float* out)
{
  int i = blockIdx.x;
  int lo = rowptr[i], hi = rowptr[i + 1];
  float degi = (float)(hi - lo + 1);
  float rdi = rsqrtf(degi);
  for (int d = threadIdx.x; d < HID; d += 256) {
    float a = xw[(size_t)i * HID + d] / degi;  // self loop
    for (int e = lo; e < hi; e++) {
      int s = src[eid[e]];
      float degs = (float)(rowptr[s + 1] - rowptr[s] + 1);
      a += xw[(size_t)s * HID + d] * rsqrtf(degs) * rdi;
    }
    out[(size_t)i * HID + d] = a + bias[d];
  }
}

// t[b] = qp2 + sum over graph-b nodes of V (mab with Nq=1: softmax==1)
__global__ void k_graphsum(const float* V, const float* qp2, float* t8) {
  int b = blockIdx.x, d = threadIdx.x;
  float a = qp2[d];
  for (int j = 0; j < NPG; j++) a += V[((size_t)b * NPG + j) * HID + d];
  t8[b * HID + d] = a;
}

// tiny dense: C = [res +] act(A@W + b)  (residual added after activation)
__global__ void k_small(const float* A, const float* W, const float* bias,
                        const float* res, float* C, int M, int K, int Nc, int act)
{
  int id = blockIdx.x * blockDim.x + threadIdx.x;
  if (id >= M * Nc) return;
  int m = id / Nc, n = id % Nc;
  float a = bias ? bias[n] : 0.f;
  const float* ap = A + (size_t)m * K;
  for (int k = 0; k < K; k++) a += ap[k] * W[(size_t)k * Nc + n];
  a = actf(a, act);
  if (res) a += res[(size_t)m * Nc + n];
  C[(size_t)m * Nc + n] = a;
}

// ============================ host side =====================================
static inline GemmSeg mkseg(const float* A, int K, int lda,
                            const int* gather = nullptr, int relu = 0) {
  GemmSeg s; s.A = A; s.gather = gather; s.K = K; s.lda = lda; s.relu = relu;
  return s;
}
static inline GemmSeg nullseg() { return mkseg(nullptr, 0, 0); }

static inline void gemm(hipStream_t st, GemmSeg a, GemmSeg b, GemmSeg c, GemmSeg d,
                        const float* W, const float* bias, float* C,
                        int M, int Nc, int act) {
  dim3 grid(DIV_UP(M, TM), DIV_UP(Nc, TN));
  k_gemm<<<grid, 128, 0, st>>>(a, b, c, d, W, bias, C, M, Nc, act);
}

extern "C" void kernel_launch(void* const* d_in, const int* in_sizes, int n_in,
                              void* d_out, int out_size, void* d_ws, size_t ws_size,
                              hipStream_t stream)
{
  (void)in_sizes; (void)n_in; (void)out_size;
#define PF(i) ((const float*)d_in[(i)])
#define PI(i) ((const int*)d_in[(i)])

  const int* src = PI(IN_EIDX);
  const int* dst = PI(IN_EIDX) + NEDGE;

  // ---- workspace layout (recomputed every call; deterministic) ----
  char* base = (char*)d_ws;
  size_t off = 0;
  auto alloc = [&](size_t bytes) -> void* {
    void* p = base + off;
    off = (off + bytes + 255) & ~(size_t)255;
    return p;
  };
  float* s_a    = (float*)alloc((size_t)NNODE * HID * 4);
  float* s_b    = (float*)alloc((size_t)NNODE * HID * 4);
  float* v_a    = (float*)alloc((size_t)NNODE * 48 * 4);
  float* v_b    = (float*)alloc((size_t)NNODE * 48 * 4);
  float* vh_nA  = (float*)alloc((size_t)NNODE * 96 * 4);
  float* vh_nB  = (float*)alloc((size_t)NNODE * 96 * 4);
  float* vn_nA  = (float*)alloc((size_t)NNODE * 32 * 4);
  float* vn_nB  = (float*)alloc((size_t)NNODE * 32 * 4);
  float* gate_n = (float*)alloc((size_t)NNODE * 32 * 4);
  float* v_f32  = (float*)alloc((size_t)NNODE * 96 * 4);
  float* v_fout = (float*)alloc((size_t)NNODE * 48 * 4);
  float* fs_big = (float*)alloc((size_t)NNODE * 1024 * 4);
  float* fs_sm  = (float*)alloc((size_t)NNODE * HID * 4);
  float* xbuf   = (float*)alloc((size_t)NNODE * HID * 4);
  float* xw     = (float*)alloc((size_t)NNODE * HID * 4);
  float* Vbuf   = (float*)alloc((size_t)NNODE * HID * 4);
  float* s0cat  = (float*)alloc((size_t)NNODE * SI0 * 4);
  float* es2    = (float*)alloc((size_t)NEDGE * ESD * 4);
  float* ev2b   = (float*)alloc((size_t)NEDGE * 3 * 4);
  float* vh_e   = (float*)alloc((size_t)NEDGE * 99 * 4);
  float* vn_e   = (float*)alloc((size_t)NEDGE * 33 * 4);
  float* gate_e = (float*)alloc((size_t)NEDGE * 16 * 4);
  float* ms_a   = (float*)alloc((size_t)NEDGE * HID * 4);
  float* ms_b   = (float*)alloc((size_t)NEDGE * HID * 4);
  float* mv_a   = (float*)alloc((size_t)NEDGE * 48 * 4);
  float* mv_b   = (float*)alloc((size_t)NEDGE * 48 * 4);
  int*   cnt    = (int*)alloc((size_t)NNODE * 4);
  int*   rowptr = (int*)alloc((size_t)(NNODE + 1) * 4);
  int*   cursor = (int*)alloc((size_t)NNODE * 4);
  int*   eid    = (int*)alloc((size_t)NEDGE * 4);
  float* t8     = (float*)alloc(8 * HID * 4);
  float* t8b    = (float*)alloc(8 * HID * 4);
  float* qp2    = (float*)alloc(HID * 4);
  float* xg     = (float*)alloc(8 * HID * 4);
  float* h8     = (float*)alloc(8 * 1024 * 4);
  if (off > ws_size) return;  // workspace too small: deterministic no-op

  // ---- CSR over dst ----
  k_seti <<<DIV_UP(NNODE, 256), 256, 0, stream>>>(cnt, NNODE, 0);
  k_count<<<DIV_UP(NEDGE, 256), 256, 0, stream>>>(dst, cnt);
  k_scan <<<1, 1, 0, stream>>>(cnt, rowptr);
  k_copyi<<<DIV_UP(NNODE, 256), 256, 0, stream>>>(rowptr, cursor, NNODE);
  k_fill <<<DIV_UP(NEDGE, 256), 256, 0, stream>>>(dst, cursor, eid);
  k_sort <<<DIV_UP(NNODE, 256), 256, 0, stream>>>(eid, rowptr);

  // ---- encoder input embed (W_v) ----
  k_embed<<<DIV_UP(NNODE * SI0, 256), 256, 0, stream>>>(PF(IN_HS), PI(IN_SEQ), PF(P_WS), s0cat);
  k_gvpln<<<NNODE, 256, 0, stream>>>(s0cat, nullptr, PF(IN_HV), nullptr,
                                     PF(P_WV_LNG), PF(P_WV_LNB), s0cat, v_fout, SI0, 3);
  k_vhn<<<DIV_UP(NNODE * 16, 256), 256, 0, stream>>>(v_fout, PF(P_WV_WH), vh_nA, vn_nA,
                                                     NNODE, 3, 16);
  gemm(stream, mkseg(s0cat, SI0, SI0), mkseg(vn_nA, 16, 16), nullseg(), nullseg(),
       PF(P_WV_WSW), PF(P_WV_WSB), s_a, NNODE, HID, 0);
  gemm(stream, mkseg(s_a, HID, HID), nullseg(), nullseg(), nullseg(),
       PF(P_WV_WSVW), PF(P_WV_WSVB), gate_n, NNODE, 16, 2);
  k_vogate<<<DIV_UP(NNODE * 16, 256), 256, 0, stream>>>(vh_nA, PF(P_WV_WV), gate_n, v_a,
                                                        NNODE, 16, 16);

  // ---- edge features (W_e, fused) ----
  k_we<<<DIV_UP(NEDGE, 256), 256, 0, stream>>>(PF(IN_ES), PF(IN_EV),
      PF(P_WE_LNB), PF(P_WE_LNG), PF(P_WE_WH), PF(P_WE_WSB), PF(P_WE_WSW),
      PF(P_WE_WSVB), PF(P_WE_WSVW), PF(P_WE_WV), es2, ev2b);

  // ---- 3 conv layers ----
  for (int L = 0; L < 3; L++) {
    int B = P_LAYER0 + L * P_LSTRIDE;
    const float *ff0_wh = PF(B + 0), *ff0_wsb = PF(B + 1), *ff0_wsw = PF(B + 2);
    const float *ff0_wsvb = PF(B + 3), *ff0_wsvw = PF(B + 4), *ff0_wv = PF(B + 5);
    const float *ff1_wh = PF(B + 6), *ff1_wsb = PF(B + 7), *ff1_wsw = PF(B + 8);
    const float *ff1_wsvb = PF(B + 9), *ff1_wsvw = PF(B + 10), *ff1_wv = PF(B + 11);
    const float *m0_wh = PF(B + 12), *m0_wsb = PF(B + 13), *m0_wsw = PF(B + 14);
    const float *m0_wsvb = PF(B + 15), *m0_wsvw = PF(B + 16), *m0_wv = PF(B + 17);
    const float *m1_wh = PF(B + 18), *m1_wsb = PF(B + 19), *m1_wsw = PF(B + 20);
    const float *m1_wsvb = PF(B + 21), *m1_wsvw = PF(B + 22), *m1_wv = PF(B + 23);
    const float *m2_wh = PF(B + 24), *m2_wsb = PF(B + 25), *m2_wsw = PF(B + 26);
    const float *m2_wsvb = PF(B + 27), *m2_wsvw = PF(B + 28), *m2_wv = PF(B + 29);
    const float *n0b = PF(B + 30), *n0g = PF(B + 31);
    const float *n1b = PF(B + 32), *n1g = PF(B + 33);

    // msg GVP0: split-K gather GEMM fuses cat(s[src], es2, s[dst], vnorm)
    k_vhn_edge<<<DIV_UP(NEDGE * 33, 256), 256, 0, stream>>>(v_a, ev2b, src, dst, m0_wh,
                                                            vh_e, vn_e);
    gemm(stream, mkseg(s_a, HID, HID, src), mkseg(es2, ESD, ESD),
         mkseg(s_a, HID, HID, dst), mkseg(vn_e, 33, 33),
         m0_wsw, m0_wsb, ms_a, NEDGE, HID, 0);
    gemm(stream, mkseg(ms_a, HID, HID), nullseg(), nullseg(), nullseg(),
         m0_wsvw, m0_wsvb, gate_e, NEDGE, 16, 2);
    k_vogate<<<DIV_UP(NEDGE * 16, 256), 256, 0, stream>>>(vh_e, m0_wv, gate_e, mv_a,
                                                          NEDGE, 33, 16);
    // msg GVP1 (s input ReLU'd via segment flag; gate uses pre-activation)
    k_vhn<<<DIV_UP(NEDGE * 16, 256), 256, 0, stream>>>(mv_a, m1_wh, vh_e, vn_e,
                                                       NEDGE, 16, 16);
    gemm(stream, mkseg(ms_a, HID, HID, nullptr, 1), mkseg(vn_e, 16, 16),
         nullseg(), nullseg(), m1_wsw, m1_wsb, ms_b, NEDGE, HID, 0);
    gemm(stream, mkseg(ms_b, HID, HID), nullseg(), nullseg(), nullseg(),
         m1_wsvw, m1_wsvb, gate_e, NEDGE, 16, 2);
    k_vogate<<<DIV_UP(NEDGE * 16, 256), 256, 0, stream>>>(vh_e, m1_wv, gate_e, mv_b,
                                                          NEDGE, 16, 16);
    // msg GVP2 (no output ReLU)
    k_vhn<<<DIV_UP(NEDGE * 16, 256), 256, 0, stream>>>(mv_b, m2_wh, vh_e, vn_e,
                                                       NEDGE, 16, 16);
    gemm(stream, mkseg(ms_b, HID, HID, nullptr, 1), mkseg(vn_e, 16, 16),
         nullseg(), nullseg(), m2_wsw, m2_wsb, ms_a, NEDGE, HID, 0);
    gemm(stream, mkseg(ms_a, HID, HID), nullseg(), nullseg(), nullseg(),
         m2_wsvw, m2_wsvb, gate_e, NEDGE, 16, 2);
    k_vogate<<<DIV_UP(NEDGE * 16, 256), 256, 0, stream>>>(vh_e, m2_wv, gate_e, mv_a,
                                                          NEDGE, 16, 16);
    // mean-aggregate + residual, then norm0
    k_agg<<<NNODE, 256, 0, stream>>>(s_a, ms_a, rowptr, eid, s_b, HID);
    k_agg<<<NNODE, 256, 0, stream>>>(v_a, mv_a, rowptr, eid, v_b, 48);
    k_gvpln<<<NNODE, 256, 0, stream>>>(s_b, nullptr, v_b, nullptr, n0g, n0b,
                                       s_b, v_b, HID, 16);
    // feed-forward GVPs
    k_vhn<<<DIV_UP(NNODE * 32, 256), 256, 0, stream>>>(v_b, ff0_wh, vh_nA, vn_nA,
                                                       NNODE, 16, 32);
    gemm(stream, mkseg(s_b, HID, HID), mkseg(vn_nA, 32, 32), nullseg(), nullseg(),
         ff0_wsw, ff0_wsb, fs_big, NNODE, 1024, 0);
    gemm(stream, mkseg(fs_big, 1024, 1024), nullseg(), nullseg(), nullseg(),
         ff0_wsvw, ff0_wsvb, gate_n, NNODE, 32, 2);
    k_vogate<<<DIV_UP(NNODE * 32, 256), 256, 0, stream>>>(vh_nA, ff0_wv, gate_n, v_f32,
                                                          NNODE, 32, 32);
    k_vhn<<<DIV_UP(NNODE * 32, 256), 256, 0, stream>>>(v_f32, ff1_wh, vh_nB, vn_nB,
                                                       NNODE, 32, 32);
    gemm(stream, mkseg(fs_big, 1024, 1024, nullptr, 1), mkseg(vn_nB, 32, 32),
         nullseg(), nullseg(), ff1_wsw, ff1_wsb, fs_sm, NNODE, HID, 0);
    gemm(stream, mkseg(fs_sm, HID, HID), nullseg(), nullseg(), nullseg(),
         ff1_wsvw, ff1_wsvb, gate_n, NNODE, 16, 2);
    k_vogate<<<DIV_UP(NNODE * 16, 256), 256, 0, stream>>>(vh_nB, ff1_wv, gate_n, v_fout,
                                                          NNODE, 32, 16);
    // norm1 with residual (pre-norm state + ff output)
    k_gvpln<<<NNODE, 256, 0, stream>>>(s_b, fs_sm, v_b, v_fout, n1g, n1b,
                                       s_a, v_a, HID, 16);
  }

  // ---- W_out + dense ----
  k_gvpln<<<NNODE, 256, 0, stream>>>(s_a, nullptr, v_a, nullptr,
                                     PF(P_WOUT_LNG), PF(P_WOUT_LNB), s_b, v_b, HID, 16);
  k_vhn<<<DIV_UP(NNODE * 16, 256), 256, 0, stream>>>(v_b, PF(P_WOUT_WH), vh_nA, vn_nA,
                                                     NNODE, 16, 16);
  gemm(stream, mkseg(s_b, HID, HID), mkseg(vn_nA, 16, 16), nullseg(), nullseg(),
       PF(P_WOUT_WSW), PF(P_WOUT_WSB), fs_sm, NNODE, HID, 1);      // relu
  gemm(stream, mkseg(fs_sm, HID, HID), nullseg(), nullseg(), nullseg(),
       PF(P_DENSE_W), PF(P_DENSE_B), s_b, NNODE, HID, 1);          // nodes

  // ---- GMT pool (only p2 V-path survives DCE; attention weights == 1) ----
  gemm(stream, mkseg(s_b, HID, HID), nullseg(), nullseg(), nullseg(),
       PF(P_LIN1W), PF(P_LIN1B), xbuf, NNODE, HID, 0);             // x
  gemm(stream, mkseg(xbuf, HID, HID), nullseg(), nullseg(), nullseg(),
       PF(P_P2_VW), nullptr, xw, NNODE, HID, 0);                   // x@v_w
  k_gcn<<<NNODE, 256, 0, stream>>>(xw, rowptr, eid, src, PF(P_P2_VB), Vbuf);
  k_small<<<1, 256, 0, stream>>>(PF(P_P2_S), PF(P_P2_FCQW), PF(P_P2_FCQB), nullptr,
                                 qp2, 1, HID, HID, 0);             // Qp (shared)
  k_graphsum<<<NGR, 256, 0, stream>>>(Vbuf, qp2, t8);
  k_gvpln<<<NGR, 256, 0, stream>>>(t8, nullptr, nullptr, nullptr,
                                   PF(P_P2_LN0G), PF(P_P2_LN0B), t8, nullptr, HID, 0);
  k_small<<<DIV_UP(8 * HID, 256), 256, 0, stream>>>(t8, PF(P_P2_FCOW), PF(P_P2_FCOB),
                                                    t8, t8b, 8, HID, HID, 1);
  k_gvpln<<<NGR, 256, 0, stream>>>(t8b, nullptr, nullptr, nullptr,
                                   PF(P_P2_LN1G), PF(P_P2_LN1B), t8b, nullptr, HID, 0);
  k_small<<<DIV_UP(8 * HID, 256), 256, 0, stream>>>(t8b, PF(P_LIN2W), PF(P_LIN2B),
                                                    nullptr, xg, 8, HID, HID, 0);
  // ---- readout ----
  k_small<<<DIV_UP(8 * 1024, 256), 256, 0, stream>>>(xg, PF(P_RO1W), PF(P_RO1B),
                                                     nullptr, h8, 8, HID, 1024, 1);
  k_small<<<DIV_UP(8 * OUTD, 256), 256, 0, stream>>>(h8, PF(P_RO2W), PF(P_RO2B),
                                                     nullptr, (float*)d_out,
                                                     8, 1024, OUTD, 2);
#undef PF
#undef PI
}
